// AttentionHead_49538152792152
// MI455X (gfx1250) — matrix operand order repaired and checked by reading, not compiled, not run on hardware
//
#include <hip/hip_runtime.h>
#include <hip/hip_bf16.h>

typedef __attribute__((ext_vector_type(16))) _Float16 v16h;
typedef __attribute__((ext_vector_type(8)))  _Float16 v8h;
typedef __attribute__((ext_vector_type(8)))  float    v8f;

#define MODEL_DIM 1024
#define HEAD_DIM  128
#define BATCH     4
#define SEQ       4096
#define ROWS      (BATCH * SEQ)          // 16384
#define SCALE     0.08838834764831845f   // 1/sqrt(128)

// ---------------------------------------------------------------------------
// Kernel 1: convert & transpose weights to f16: Wt[mat][d][m], fold softmax
// scale into Wq so the score matmul needs no extra multiply.
// ---------------------------------------------------------------------------
__global__ __launch_bounds__(256) void wt_convert(
    const float* __restrict__ Wq, const float* __restrict__ Wk,
    const float* __restrict__ Wv, _Float16* __restrict__ Wt) {
  int idx = blockIdx.x * 256 + threadIdx.x;
  if (idx >= 3 * MODEL_DIM * HEAD_DIM) return;
  int mat = idx / (MODEL_DIM * HEAD_DIM);
  int e   = idx % (MODEL_DIM * HEAD_DIM);
  int m   = e / HEAD_DIM;
  int d   = e % HEAD_DIM;
  const float* W = (mat == 0) ? Wq : (mat == 1) ? Wk : Wv;
  float w = W[e];
  if (mat == 0) w *= SCALE;
  Wt[(size_t)mat * (MODEL_DIM * HEAD_DIM) + d * MODEL_DIM + m] = (_Float16)w;
}

// ---------------------------------------------------------------------------
// Kernel 2: QKV projection with WMMA f16->f32.
// One wave computes one 16x128 output tile of one of Q/K/V.
// All 8 B-fragments are preloaded per k-chunk so the WMMA stream overlaps the
// next fragment loads instead of stalling on s_wait_loadcnt 0 per WMMA.
// ---------------------------------------------------------------------------
__global__ __launch_bounds__(256) void qkv_proj(
    const float*    __restrict__ x,    // [16384][1024] f32
    const _Float16* __restrict__ Wt,   // [3][128][1024] f16 (transposed)
    const float*    __restrict__ bq,
    const float*    __restrict__ bk,
    const float*    __restrict__ bv,
    _Float16*       __restrict__ qkv)  // [3][16384][128] f16
{
  const int lane = threadIdx.x & 31;
  const int wave = threadIdx.x >> 5;
  const int wgid = blockIdx.x * 8 + wave;       // 0..3071
  const int mat  = wgid / (ROWS / 16);          // 0..2
  const int rt   = wgid % (ROWS / 16);
  const int r0   = rt * 16;
  const int hi   = lane >> 4;                   // which 16-lane half
  const int ln   = lane & 15;

  const float*    bias = (mat == 0) ? bq : (mat == 1) ? bk : bv;
  const _Float16* W    = Wt + (size_t)mat * (MODEL_DIM * HEAD_DIM);
  _Float16*       out  = qkv + (size_t)mat * (ROWS * HEAD_DIM);

  v8f c[8] = {};

  for (int mc = 0; mc < MODEL_DIM / 32; ++mc) {
    const int m0 = mc * 32;
    // A fragment (16x32 f16): lane holds row r0+ln;
    // K-chunks: lanes 0-15 -> K 0-15, lanes 16-31 -> K 16-31.
    const float* xr = x + (size_t)(r0 + ln) * MODEL_DIM + m0;
    v16h a;
#pragma unroll
    for (int j = 0; j < 8; ++j) a[j]     = (_Float16)xr[16 * hi + j];
#pragma unroll
    for (int j = 0; j < 8; ++j) a[8 + j] = (_Float16)xr[16 * hi + 8 + j];

    // Preload all 8 B fragments (one load clause, staggered waits).
    v16h bfrag[8];
#pragma unroll
    for (int nt = 0; nt < 8; ++nt) {
      // B fragment (32x16 f16): lane holds col d = nt*16+ln,
      // K rows m0 + 16*hi + 0..15 (contiguous in Wt).
      const _Float16* wp = W + (size_t)(nt * 16 + ln) * MODEL_DIM + m0 + 16 * hi;
      v8h t0 = *(const v8h*)wp;
      v8h t1 = *(const v8h*)(wp + 8);
#pragma unroll
      for (int j = 0; j < 8; ++j) { bfrag[nt][j] = t0[j]; bfrag[nt][8 + j] = t1[j]; }
    }
#pragma unroll
    for (int nt = 0; nt < 8; ++nt) {
      c[nt] = __builtin_amdgcn_wmma_f32_16x16x32_f16(
          false, a, false, bfrag[nt], (short)0, c[nt], false, false);
    }
  }

  const float sc = (mat == 0) ? SCALE : 1.0f;
#pragma unroll
  for (int nt = 0; nt < 8; ++nt) {
    const int col = nt * 16 + ln;
    const float bb = bias[col] * sc;
#pragma unroll
    for (int r = 0; r < 8; ++r) {
      const int row = r0 + r + 8 * hi;   // C layout: M = r + 8*(lane/16)
      out[(size_t)row * HEAD_DIM + col] = (_Float16)(c[nt][r] + bb);
    }
  }
}

// ---------------------------------------------------------------------------
// Kernel 3: flash attention. 1 wave = 16 query rows; 8 waves/WG; KV tile = 32.
// K staged row-major in LDS (score B-frags contiguous), V staged transposed
// (PV B-frags contiguous). P re-layout C->A via per-wave LDS scratch.
// Global->register loads for tile kt+1 are issued before the compute of tile
// kt (software pipeline), so HBM/L2 latency hides under the 16 WMMAs.
// ---------------------------------------------------------------------------
__global__ __launch_bounds__(256) void flash_attn(
    const _Float16* __restrict__ Qh,   // [B*S][128] prescaled by 1/sqrt(128)
    const _Float16* __restrict__ Kh,
    const _Float16* __restrict__ Vh,
    float*          __restrict__ out)  // [B*S][128] f32
{
  __shared__ alignas(16) _Float16 k_lds[32][HEAD_DIM];   // 8 KB
  __shared__ alignas(16) _Float16 vt_lds[HEAD_DIM][32];  // 8 KB (transposed)
  __shared__ alignas(16) _Float16 p_lds[8][16][32];      // 8 KB, per-wave

  const int lane = threadIdx.x & 31;
  const int wave = threadIdx.x >> 5;
  const int hi   = lane >> 4;
  const int ln   = lane & 15;
  const int b    = blockIdx.x >> 5;                     // 0..3
  const int q0   = (blockIdx.x & 31) * 128 + wave * 16; // 0..4080

  const _Float16* Qb = Qh + (size_t)b * SEQ * HEAD_DIM;
  const _Float16* Kb = Kh + (size_t)b * SEQ * HEAD_DIM;
  const _Float16* Vb = Vh + (size_t)b * SEQ * HEAD_DIM;

  // Per-thread staging role: kv row srow, 16-half chunk soff.
  const int srow = threadIdx.x >> 3;
  const int soff = (threadIdx.x & 7) * 16;

  // Load the 16x128 Q tile as four 16x32 A-fragments (held for whole loop).
  v16h aq[4];
#pragma unroll
  for (int dc = 0; dc < 4; ++dc) {
    const _Float16* qp = Qb + (size_t)(q0 + ln) * HEAD_DIM + dc * 32;
    v8h t0 = *(const v8h*)(qp + 16 * hi);
    v8h t1 = *(const v8h*)(qp + 16 * hi + 8);
#pragma unroll
    for (int j = 0; j < 8; ++j) { aq[dc][j] = t0[j]; aq[dc][8 + j] = t1[j]; }
  }

  v8f o[8] = {};
  float mrow[8], lrow[8];
#pragma unroll
  for (int r = 0; r < 8; ++r) { mrow[r] = -1e30f; lrow[r] = 0.0f; }

  // Pipeline prologue: fetch KV tile 0 into registers.
  v8h kr0, kr1, vr0, vr1;
  {
    const _Float16* kp = Kb + (size_t)srow * HEAD_DIM + soff;
    kr0 = *(const v8h*)kp; kr1 = *(const v8h*)(kp + 8);
    const _Float16* vp = Vb + (size_t)srow * HEAD_DIM + soff;
    vr0 = *(const v8h*)vp; vr1 = *(const v8h*)(vp + 8);
  }

  for (int kt = 0; kt < SEQ / 32; ++kt) {
    __syncthreads();  // previous iteration's LDS readers are done
    // Commit staged registers for tile kt to LDS.
    *(v8h*)&k_lds[srow][soff]     = kr0;
    *(v8h*)&k_lds[srow][soff + 8] = kr1;
#pragma unroll
    for (int j = 0; j < 8; ++j) vt_lds[soff + j][srow]     = vr0[j];
#pragma unroll
    for (int j = 0; j < 8; ++j) vt_lds[soff + 8 + j][srow] = vr1[j];
    __syncthreads();

    // Issue global loads for tile kt+1 now; they drain under the WMMAs below.
    if (kt + 1 < SEQ / 32) {
      const int sn = (kt + 1) * 32 + srow;
      const _Float16* kp = Kb + (size_t)sn * HEAD_DIM + soff;
      kr0 = *(const v8h*)kp; kr1 = *(const v8h*)(kp + 8);
      const _Float16* vp = Vb + (size_t)sn * HEAD_DIM + soff;
      vr0 = *(const v8h*)vp; vr1 = *(const v8h*)(vp + 8);
    }

    // Scores: S[16q x 32kv] = Q (prescaled) . K^T, two 16x16 C tiles.
    v8f sc0 = {}, sc1 = {};
#pragma unroll
    for (int dc = 0; dc < 4; ++dc) {
      // B frag: col = rel kv (16*tile + ln), K rows d = dc*32 + 16*hi + 0..15
      const _Float16* kp0 = &k_lds[ln][dc * 32 + 16 * hi];
      const _Float16* kp1 = &k_lds[16 + ln][dc * 32 + 16 * hi];
      v8h a0 = *(const v8h*)kp0, a1 = *(const v8h*)(kp0 + 8);
      v8h b0 = *(const v8h*)kp1, b1 = *(const v8h*)(kp1 + 8);
      v16h bk0, bk1;
#pragma unroll
      for (int j = 0; j < 8; ++j) {
        bk0[j] = a0[j]; bk0[8 + j] = a1[j];
        bk1[j] = b0[j]; bk1[8 + j] = b1[j];
      }
      sc0 = __builtin_amdgcn_wmma_f32_16x16x32_f16(
          false, aq[dc], false, bk0, (short)0, sc0, false, false);
      sc1 = __builtin_amdgcn_wmma_f32_16x16x32_f16(
          false, aq[dc], false, bk1, (short)0, sc1, false, false);
    }

    // Online softmax (rows replicated across each 16-lane half of the wave).
#pragma unroll
    for (int r = 0; r < 8; ++r) {
      float s0v = sc0[r], s1v = sc1[r];
      float rm = fmaxf(s0v, s1v);
#pragma unroll
      for (int msk = 8; msk >= 1; msk >>= 1)
        rm = fmaxf(rm, __shfl_xor(rm, msk, 32));
      const float nm    = fmaxf(mrow[r], rm);
      const float alpha = __expf(mrow[r] - nm);
      const float p0    = __expf(s0v - nm);
      const float p1    = __expf(s1v - nm);
      float rs = p0 + p1;
#pragma unroll
      for (int msk = 8; msk >= 1; msk >>= 1)
        rs += __shfl_xor(rs, msk, 32);
      mrow[r] = nm;
      lrow[r] = lrow[r] * alpha + rs;
      const int prow = r + 8 * hi;   // C layout row
      p_lds[wave][prow][ln]      = (_Float16)p0;
      p_lds[wave][prow][16 + ln] = (_Float16)p1;
#pragma unroll
      for (int nt = 0; nt < 8; ++nt) o[nt][r] *= alpha;
    }

    // Re-load P tile as one 16x32 A-fragment (per-wave LDS, no barrier needed).
    v16h ap;
    {
      const _Float16* pp = &p_lds[wave][ln][0];
      v8h t0 = *(const v8h*)(pp + 16 * hi);
      v8h t1 = *(const v8h*)(pp + 16 * hi + 8);
#pragma unroll
      for (int j = 0; j < 8; ++j) { ap[j] = t0[j]; ap[8 + j] = t1[j]; }
    }

    // O += P . V  (8 n-tiles of 16 head dims)
#pragma unroll
    for (int nt = 0; nt < 8; ++nt) {
      const _Float16* vp = &vt_lds[nt * 16 + ln][16 * hi];
      v8h t0 = *(const v8h*)vp, t1 = *(const v8h*)(vp + 8);
      v16h bvf;
#pragma unroll
      for (int j = 0; j < 8; ++j) { bvf[j] = t0[j]; bvf[8 + j] = t1[j]; }
      o[nt] = __builtin_amdgcn_wmma_f32_16x16x32_f16(
          false, ap, false, bvf, (short)0, o[nt], false, false);
    }
  }

  // Normalize and store f32 output.
#pragma unroll
  for (int r = 0; r < 8; ++r) {
    const float inv = 1.0f / lrow[r];
    const int row = q0 + r + 8 * hi;
    float* op = out + ((size_t)b * SEQ + row) * HEAD_DIM;
#pragma unroll
    for (int nt = 0; nt < 8; ++nt) op[nt * 16 + ln] = o[nt][r] * inv;
  }
}

// ---------------------------------------------------------------------------
extern "C" void kernel_launch(void* const* d_in, const int* in_sizes, int n_in,
                              void* d_out, int out_size, void* d_ws, size_t ws_size,
                              hipStream_t stream) {
  const float* x  = (const float*)d_in[0];
  const float* Wq = (const float*)d_in[1];
  const float* bq = (const float*)d_in[2];
  const float* Wk = (const float*)d_in[3];
  const float* bk = (const float*)d_in[4];
  const float* Wv = (const float*)d_in[5];
  const float* bv = (const float*)d_in[6];
  float* out = (float*)d_out;

  // Workspace: [3][16384][128] f16 QKV (12 MB) + [3][128][1024] f16 Wt (768 KB)
  _Float16* qkv = (_Float16*)d_ws;
  _Float16* Wt  = qkv + (size_t)3 * ROWS * HEAD_DIM;

  wt_convert<<<(3 * MODEL_DIM * HEAD_DIM + 255) / 256, 256, 0, stream>>>(
      Wq, Wk, Wv, Wt);
  qkv_proj<<<384, 256, 0, stream>>>(x, Wt, bq, bk, bv, qkv);
  flash_attn<<<128, 256, 0, stream>>>(qkv,
                                      qkv + (size_t)ROWS * HEAD_DIM,
                                      qkv + (size_t)2 * ROWS * HEAD_DIM,
                                      out);
}
